// SparseMoE_8804682956825
// MI455X (gfx1250) — matrix operand order repaired and checked by reading, not compiled, not run on hardware
//
#include <hip/hip_runtime.h>
#include <hip/hip_bf16.h>

// ---------------------------------------------------------------------------
// Sparse top-2 MoE for MI455X (gfx1250, wave32, WMMA), N=8192, D=1024,
// H=4096, O=1024, E=8, top-k=2.
//
// Pipeline (all on one stream):
//   0) zero out + counters
//   1) transpose+convert W1,W2 -> bf16 K-major (WT[col][k]) in d_ws (once per
//      launch; amortized over ~128 consumer blocks per expert). ~128.6 MB ws.
//   2) gating: top-2 + softmax, scatter tokens into per-expert buckets
//   3) fused expert MLP: per 16-token tile, x tile (32KB) + full hidden tile
//      (128KB) in LDS; B fragments stream via global_load_b128 from bf16
//      K-major weights; A fragments are 2x ds_load_b128; software-pipelined
//      (depth 1) barrier-free K loop, 2 WMMA per step.
// ---------------------------------------------------------------------------

#define N_TOK 8192
#define DIM_D 1024
#define DIM_H 4096
#define DIM_O 1024
#define N_EXP 8
#define TM    16      // token rows per block

typedef __bf16 bf16;
typedef __attribute__((ext_vector_type(16))) __bf16 v16bf;
typedef __attribute__((ext_vector_type(8)))  float  v8f;

union Frag16 { v16bf v; bf16 e[16]; uint4 q[2]; };
union Frag8f { v8f  v; float e[8]; };

// ---------------------------------------------------------------------------
// Kernel 0: zero output + expert counters (buffers are poisoned by harness).
// ---------------------------------------------------------------------------
__global__ void moe_zero_kernel(float* __restrict__ out, int* __restrict__ counts,
                                size_t n) {
    size_t i = (size_t)blockIdx.x * blockDim.x + threadIdx.x;
    size_t stride = (size_t)gridDim.x * blockDim.x;
    for (; i < n; i += stride) out[i] = 0.0f;
    if (blockIdx.x == 0 && threadIdx.x < N_EXP) counts[threadIdx.x] = 0;
}

// ---------------------------------------------------------------------------
// Kernel 1: transpose + fp32->bf16. W:[E][R][C] -> WT:[E][C][R], 64x64 tiles
// via LDS. Coalesced float4 reads, coalesced b64 bf16 writes.
// ---------------------------------------------------------------------------
__global__ __launch_bounds__(256)
void moe_transpose_kernel(const float* __restrict__ W, bf16* __restrict__ WT,
                          int R, int C) {
    const int e  = blockIdx.z;
    const int r0 = blockIdx.y * 64;
    const int c0 = blockIdx.x * 64;
    const float* We  = W  + (size_t)e * R * C;
    bf16*        WTe = WT + (size_t)e * R * C;
    __shared__ bf16 tile[64 * 72];                 // +8 pad vs banks
    const int tid = threadIdx.x;
#pragma unroll
    for (int q = 0; q < 4; ++q) {
        int idx4  = tid + 256 * q;                 // 1024 float4 chunks
        int inrow = idx4 >> 4;
        int incol = (idx4 & 15) << 2;
        const float4 w = *(const float4*)(We + (size_t)(r0 + inrow) * C + c0 + incol);
        tile[(incol + 0) * 72 + inrow] = (bf16)w.x;
        tile[(incol + 1) * 72 + inrow] = (bf16)w.y;
        tile[(incol + 2) * 72 + inrow] = (bf16)w.z;
        tile[(incol + 3) * 72 + inrow] = (bf16)w.w;
    }
    __syncthreads();
#pragma unroll
    for (int q = 0; q < 4; ++q) {
        int idx4   = tid + 256 * q;
        int outrow = idx4 >> 4;                    // c within tile
        int outcol = (idx4 & 15) << 2;             // r within tile
        union { bf16 h[4]; uint2 u; } p;
        p.h[0] = tile[outrow * 72 + outcol + 0];
        p.h[1] = tile[outrow * 72 + outcol + 1];
        p.h[2] = tile[outrow * 72 + outcol + 2];
        p.h[3] = tile[outrow * 72 + outcol + 3];
        *(uint2*)(WTe + (size_t)(c0 + outrow) * R + r0 + outcol) = p.u;
    }
}

// ---------------------------------------------------------------------------
// Kernel 2: gating. One wave per token: 8 dots over D=1024, butterfly reduce,
// top-2 + 2-way softmax, atomic scatter into per-expert buckets.
// ---------------------------------------------------------------------------
__global__ void moe_gate_kernel(const float* __restrict__ x,
                                const float* __restrict__ Wg,
                                const float* __restrict__ bg,
                                int* __restrict__ counts,
                                int* __restrict__ btok,
                                float* __restrict__ bp) {
    const int wave = threadIdx.x >> 5;
    const int lane = threadIdx.x & 31;
    const int token = blockIdx.x * 8 + wave;

    float acc[N_EXP];
#pragma unroll
    for (int e = 0; e < N_EXP; ++e) acc[e] = 0.0f;

    const float* xr = x + (size_t)token * DIM_D;
    for (int k = lane; k < DIM_D; k += 32) {
        float xv = xr[k];
#pragma unroll
        for (int e = 0; e < N_EXP; ++e) acc[e] += xv * Wg[e * DIM_D + k];
    }
#pragma unroll
    for (int e = 0; e < N_EXP; ++e) {
        float v = acc[e];
        for (int off = 16; off > 0; off >>= 1) v += __shfl_xor(v, off);
        acc[e] = v + bg[e];
    }
    if (lane == 0) {
        int e1 = 0; float v1 = acc[0];
#pragma unroll
        for (int e = 1; e < N_EXP; ++e)
            if (acc[e] > v1) { v1 = acc[e]; e1 = e; }
        int e2 = (e1 == 0) ? 1 : 0; float v2 = acc[e2];
#pragma unroll
        for (int e = 0; e < N_EXP; ++e)
            if (e != e1 && acc[e] > v2) { v2 = acc[e]; e2 = e; }
        float eb = __expf(v2 - v1);             // stable: v1 >= v2
        float inv = 1.0f / (1.0f + eb);
        int pos1 = atomicAdd(&counts[e1], 1);
        btok[e1 * N_TOK + pos1] = token; bp[e1 * N_TOK + pos1] = inv;
        int pos2 = atomicAdd(&counts[e2], 1);
        btok[e2 * N_TOK + pos2] = token; bp[e2 * N_TOK + pos2] = eb * inv;
    }
}

// ---------------------------------------------------------------------------
// Software-pipelined (depth-1) K loop: acc0/acc1 += A[16 x 32*KSTEPS] * B
// for two adjacent 16-col tiles. A from LDS row (2x ds_load_b128), B from
// bf16 K-major global (2x global_load_b128 per tile). Next step's fragments
// are loaded before this step's WMMAs; last step peeled. No barriers.
// ---------------------------------------------------------------------------
template <int KSTEPS>
__device__ __forceinline__ void gemm_kloop(const bf16* __restrict__ arow,
                                           const bf16* __restrict__ bbase0,
                                           const bf16* __restrict__ bbase1,
                                           Frag8f& acc0, Frag8f& acc1) {
    Frag16 a, b0, b1f;
    a.q[0]   = *(const uint4*)(arow);
    a.q[1]   = *(const uint4*)(arow + 8);
    b0.q[0]  = *(const uint4*)(bbase0);
    b0.q[1]  = *(const uint4*)(bbase0 + 8);
    b1f.q[0] = *(const uint4*)(bbase1);
    b1f.q[1] = *(const uint4*)(bbase1 + 8);

#pragma unroll 2
    for (int ks = 0; ks < KSTEPS - 1; ++ks) {
        const int kn = (ks + 1) * 32;
        if ((ks & 1) == 0)   // constant-folds under unroll-2
            __builtin_prefetch(bbase0 + kn + 512, 0, 3);   // ~1KB ahead, near$
        Frag16 an, b0n, b1n;
        an.q[0]  = *(const uint4*)(arow + kn);
        an.q[1]  = *(const uint4*)(arow + kn + 8);
        b0n.q[0] = *(const uint4*)(bbase0 + kn);
        b0n.q[1] = *(const uint4*)(bbase0 + kn + 8);
        b1n.q[0] = *(const uint4*)(bbase1 + kn);
        b1n.q[1] = *(const uint4*)(bbase1 + kn + 8);
        acc0.v = __builtin_amdgcn_wmma_f32_16x16x32_bf16(
            false, a.v, false, b0.v, (short)0, acc0.v, false, false);
        acc1.v = __builtin_amdgcn_wmma_f32_16x16x32_bf16(
            false, a.v, false, b1f.v, (short)0, acc1.v, false, false);
        a = an; b0 = b0n; b1f = b1n;
    }
    acc0.v = __builtin_amdgcn_wmma_f32_16x16x32_bf16(
        false, a.v, false, b0.v, (short)0, acc0.v, false, false);
    acc1.v = __builtin_amdgcn_wmma_f32_16x16x32_bf16(
        false, a.v, false, b1f.v, (short)0, acc1.v, false, false);
}

// ---------------------------------------------------------------------------
// Kernel 3: fused expert MLP. grid = (512 tiles, 8 experts), 256 threads
// (8 waves). Each wave owns independent 32-col output groups.
// ---------------------------------------------------------------------------
__global__ __launch_bounds__(256)
void moe_expert_kernel(const float* __restrict__ x,
                       const bf16* __restrict__ W1T,   // [E][H][D] bf16
                       const float* __restrict__ b1,
                       const bf16* __restrict__ W2T,   // [E][O][H] bf16
                       const float* __restrict__ b2,
                       const int* __restrict__ counts,
                       const int* __restrict__ btok,
                       const float* __restrict__ bp,
                       float* __restrict__ out) {
    const int e = blockIdx.y;
    const int t = blockIdx.x;
    const int cnt = counts[e];
    if (t * TM >= cnt) return;               // uniform across block: safe

    extern __shared__ char smem_raw[];
    float* p_s   = (float*)smem_raw;                 // 16 floats
    int*   tok_s = (int*)(p_s + TM);                 // 16 ints
    bf16*  lds_x = (bf16*)(tok_s + TM);              // TM*DIM_D (32 KB)
    bf16*  lds_h = lds_x + TM * DIM_D;               // TM*DIM_H (128 KB)

    const int tid  = threadIdx.x;
    const int wave = tid >> 5;
    const int lane = tid & 31;

    if (tid < TM) {
        int r = t * TM + tid;
        bool valid = r < cnt;
        int rr = valid ? r : (cnt - 1);
        tok_s[tid] = btok[e * N_TOK + rr];
        p_s[tid]   = valid ? bp[e * N_TOK + rr] : 0.0f;
    }
    __syncthreads();

    // stage x tile fp32 -> bf16 (float4 loads, b64 LDS stores)
    for (int idx = tid; idx < TM * (DIM_D / 4); idx += 256) {
        int row = idx >> 8;                  // DIM_D/4 = 256 chunks per row
        int c4  = (idx & 255) << 2;
        const float4 xv = *(const float4*)(x + (size_t)tok_s[row] * DIM_D + c4);
        union { bf16 h[4]; uint2 u; } p;
        p.h[0] = (bf16)xv.x; p.h[1] = (bf16)xv.y;
        p.h[2] = (bf16)xv.z; p.h[3] = (bf16)xv.w;
        *(uint2*)(lds_x + row * DIM_D + c4) = p.u;
    }
    __syncthreads();

    // Fragment lane roles (wave32 bf16; this code's contiguous k order, not the ISA table order):
    //  A 16x32: row m = lane&15; VGPR0..7 hold K=kbA+0..15 contiguous
    //           -> two b128 from a row-major LDS row.
    //  B 32x16: col n = lane&15; K=kbB+0..15 contiguous in K-major weights
    //           -> two b128 straight from global.
    //  C/D:     col n = lane&15; VGPR r = row mrow0+r.
    const int m     = lane & 15;
    const int kbA   = (lane >= 16) ? 16 : 0;
    const int kbB   = (lane >= 16) ? 16 : 0;
    const int mrow0 = (lane >= 16) ? 8 : 0;

    const bf16* W1e = W1T + (size_t)e * DIM_H * DIM_D;
    const bf16* W2e = W2T + (size_t)e * DIM_O * DIM_H;

    // ---------------- Phase 1: h = relu(x @ W1 + b1) ----------------
    for (int hg = wave; hg < DIM_H / 32; hg += 8) {
        Frag8f acc0, acc1;
#pragma unroll
        for (int r = 0; r < 8; ++r) { acc0.e[r] = 0.0f; acc1.e[r] = 0.0f; }

        const bf16* bbase0 = W1e + (size_t)(hg * 32 + m) * DIM_D + kbB;
        gemm_kloop<DIM_D / 32>(lds_x + m * DIM_D + kbA,
                               bbase0, bbase0 + (size_t)16 * DIM_D, acc0, acc1);

#pragma unroll
        for (int j = 0; j < 2; ++j) {
            const Frag8f& A = j ? acc1 : acc0;
            const int hcol = hg * 32 + (j << 4) + m;
            const float bias = b1[e * DIM_H + hcol];
#pragma unroll
            for (int r = 0; r < 8; ++r) {
                float vv = A.e[r] + bias;
                vv = vv > 0.0f ? vv : 0.0f;
                lds_h[(mrow0 + r) * DIM_H + hcol] = (bf16)vv;
            }
        }
    }

    __syncthreads();   // hidden tile complete

    // ---------------- Phase 2: out += p * (h @ W2 + b2) ----------------
    for (int og = wave; og < DIM_O / 32; og += 8) {
        Frag8f acc0, acc1;
#pragma unroll
        for (int r = 0; r < 8; ++r) { acc0.e[r] = 0.0f; acc1.e[r] = 0.0f; }

        const bf16* bbase0 = W2e + (size_t)(og * 32 + m) * DIM_H + kbB;
        gemm_kloop<DIM_H / 32>(lds_h + m * DIM_H + kbA,
                               bbase0, bbase0 + (size_t)16 * DIM_H, acc0, acc1);

#pragma unroll
        for (int j = 0; j < 2; ++j) {
            const Frag8f& A = j ? acc1 : acc0;
            const int ocol = og * 32 + (j << 4) + m;
            const float bias = b2[e * DIM_O + ocol];
#pragma unroll
            for (int r = 0; r < 8; ++r) {
                const int mr = mrow0 + r;
                const float vv = (A.e[r] + bias) * p_s[mr];
                atomicAdd(out + (size_t)tok_s[mr] * DIM_O + ocol, vv);
            }
        }
    }
}

// ---------------------------------------------------------------------------
// Launch. Workspace: 0.5 MB routing buckets + 128 MB bf16 K-major weights.
// ---------------------------------------------------------------------------
extern "C" void kernel_launch(void* const* d_in, const int* in_sizes, int n_in,
                              void* d_out, int out_size, void* d_ws, size_t ws_size,
                              hipStream_t stream) {
    (void)in_sizes; (void)n_in; (void)out_size; (void)ws_size;
    const float* x  = (const float*)d_in[0];
    const float* Wg = (const float*)d_in[1];
    const float* bg = (const float*)d_in[2];
    const float* W1 = (const float*)d_in[3];
    const float* b1 = (const float*)d_in[4];
    const float* W2 = (const float*)d_in[5];
    const float* b2 = (const float*)d_in[6];
    float* out = (float*)d_out;

    // workspace layout
    char* ws = (char*)d_ws;
    size_t off = 0;
    int*   counts = (int*)(ws + off); off += 256;
    int*   btok   = (int*)(ws + off); off += sizeof(int)   * (size_t)N_EXP * N_TOK;
    float* bp     = (float*)(ws + off); off += sizeof(float) * (size_t)N_EXP * N_TOK;
    off = (off + 255) & ~(size_t)255;
    bf16* W1T = (bf16*)(ws + off); off += sizeof(bf16) * (size_t)N_EXP * DIM_D * DIM_H;
    bf16* W2T = (bf16*)(ws + off);

    moe_zero_kernel<<<2048, 256, 0, stream>>>(out, counts,
                                              (size_t)N_TOK * DIM_O);
    // W1: [E][D][H] -> [E][H][D];  W2: [E][H][O] -> [E][O][H]
    moe_transpose_kernel<<<dim3(DIM_H / 64, DIM_D / 64, N_EXP), 256, 0, stream>>>(
        W1, W1T, DIM_D, DIM_H);
    moe_transpose_kernel<<<dim3(DIM_O / 64, DIM_H / 64, N_EXP), 256, 0, stream>>>(
        W2, W2T, DIM_H, DIM_O);
    moe_gate_kernel<<<N_TOK / 8, 256, 0, stream>>>(x, Wg, bg, counts, btok, bp);

    const size_t smem = (size_t)(TM * DIM_D + TM * DIM_H) * sizeof(bf16)
                        + TM * (sizeof(float) + sizeof(int));
    moe_expert_kernel<<<dim3(N_TOK / TM, N_EXP), 256, smem, stream>>>(
        x, W1T, b1, W2T, b2, counts, btok, bp, out);
}
